// minSAD_FFT_Affine_Registration_Layer_76020921139535
// MI455X (gfx1250) — compile-verified, matrix operand order checked by reading, not run on hardware
//
#include <hip/hip_runtime.h>
#include <hip/hip_bf16.h>
#include <math.h>

#define PI_F 3.14159265358979323846f
#define NDIM 448
#define PIX (448 * 448)

typedef __attribute__((ext_vector_type(16))) _Float16 v16h;
typedef __attribute__((ext_vector_type(4)))  _Float16 v4h;
typedef __attribute__((ext_vector_type(8)))  float    v8f;

// ---------------------------------------------------------------------------
// Build DFT matrices:  F[w,k] = e^{-2pi i wk/N},  G = F^{-1}[k,w] = (1/N) e^{+2pi i kw/N}
// ---------------------------------------------------------------------------
__global__ void k_build_dft(float* __restrict__ Fr, float* __restrict__ Fi,
                            float* __restrict__ Gr, float* __restrict__ Gi) {
    int idx = blockIdx.x * blockDim.x + threadIdx.x;
    if (idx >= PIX) return;
    int r = idx / NDIM, c = idx % NDIM;
    int ph = (int)(((long long)r * (long long)c) % NDIM);
    float ang = -2.0f * PI_F * (float)ph * (1.0f / NDIM);
    float cs = cosf(ang), sn = sinf(ang);
    Fr[idx] = cs;
    Fi[idx] = sn;
    Gr[idx] = cs * (1.0f / NDIM);
    Gi[idx] = -sn * (1.0f / NDIM);
}

// ---------------------------------------------------------------------------
// Per-angle shear parameters + zero slab.
// ---------------------------------------------------------------------------
__global__ void k_init(float* __restrict__ prot_a, float* __restrict__ prot_b,
                       float* __restrict__ zer) {
    int t = threadIdx.x;
    if (t < 21) {
        float theta = (float)(t - 10) * (PI_F / 180.0f);
        prot_a[t] = -tanf(0.5f * theta);
        prot_b[t] = sinf(theta);
    }
    if (t >= 32 && t < 64) zer[t - 32] = 0.0f;
}

// ---------------------------------------------------------------------------
// Batched complex GEMM via WMMA, compile-time specialized on operand shape:
//   AI: A has imag;  BI: B has imag;  CI: produce imag(C)
//   Cr = Ar*Br - Ai*Bi ;  Ci = Ar*Bi + Ai*Br
// Block = 256 thr = 8 waves; block tile M64 x N64; wave tile 16x32; K step 32.
// LDS staged in WMMA operand order (lane = k[3]*16 | row, e = k[4]<<3|k[2:1]<<1|k[0])
// so consume is two ds_load_b128 per operand. Imag(A) is pre-negated at staging
// when its only use is the Cr cross term (f16 WMMA has no A/B NEG per ISA).
// ---------------------------------------------------------------------------
template <bool AI, bool BI, bool CI>
__global__ __launch_bounds__(256)
void k_gemm_cplx(const float* __restrict__ Ar, const float* __restrict__ Ai, long sA,
                 const float* __restrict__ Br, const float* __restrict__ Bi, long sB,
                 float* __restrict__ Cr, float* __restrict__ Ci, long sC) {
    const int K = NDIM, N = NDIM;
    const long b = blockIdx.z;
    Ar += b * sA; if (AI) Ai += b * sA;
    Br += b * sB; if (BI) Bi += b * sB;
    Cr += b * sC; if (CI) Ci += b * sC;

    // operand-order LDS: [subtile][lane][element]; imag slabs collapse when unused
    __shared__ __align__(32) _Float16 pA [4][32][16];
    __shared__ __align__(32) _Float16 pAi[AI ? 4 : 1][32][16];
    __shared__ __align__(32) _Float16 pB [4][32][16];
    __shared__ __align__(32) _Float16 pBi[BI ? 4 : 1][32][16];

    const int tid  = threadIdx.x;
    const int lane = tid & 31;
    const int wv   = tid >> 5;           // 0..7
    const int wm   = wv >> 1;            // 0..3  (M sub-tile, 16 rows)
    const int wn   = wv & 1;             // 0..1  (pair of N sub-tiles)
    const int hh   = lane >> 4;
    const int l15  = lane & 15;

    const int mBase = blockIdx.y * 64;
    const int nBase = blockIdx.x * 64;

    // stage -Ai when the only consumer is the Cr cross term
    constexpr float aSign = (AI && BI && !CI) ? -1.0f : 1.0f;

    v8f cr0 = {}, cr1 = {}, ci0 = {}, ci1 = {};

    for (int k0 = 0; k0 < K; k0 += 32) {
        // ---- stage A (64 rows x 32 k): float4 loads, contiguous v4h LDS writes ----
        for (int i = tid * 4; i < 64 * 32; i += 256 * 4) {
            int m = i >> 5;              // 0..63
            int k = i & 31;              // multiple of 4
            size_t g = (size_t)(mBase + m) * K + (k0 + k);
            int ln = (((k >> 3) & 1) << 4) | (m & 15);
            int eb = ((k >> 4) << 3) | (((k >> 1) & 3) << 1);
            float4 f = *(const float4*)(Ar + g);
            v4h hv = {(_Float16)f.x, (_Float16)f.y, (_Float16)f.z, (_Float16)f.w};
            *(v4h*)&pA[m >> 4][ln][eb] = hv;
            if constexpr (AI) {
                float4 q = *(const float4*)(Ai + g);
                v4h hq = {(_Float16)(aSign * q.x), (_Float16)(aSign * q.y),
                          (_Float16)(aSign * q.z), (_Float16)(aSign * q.w)};
                *(v4h*)&pAi[m >> 4][ln][eb] = hq;
            }
        }
        // ---- stage B (32 k x 64 n): float4 loads, 4 lane-strided b16 writes ----
        for (int i = tid * 4; i < 32 * 64; i += 256 * 4) {
            int k = i >> 6;              // 0..31
            int n = i & 63;              // multiple of 4
            size_t g = (size_t)(k0 + k) * N + (nBase + n);
            int e  = ((k >> 4) << 3) | (((k >> 1) & 3) << 1) | (k & 1);
            int lb = (((k >> 3) & 1) << 4) | (n & 15);
            int tb = n >> 4;
            float4 f = *(const float4*)(Br + g);
            pB[tb][lb + 0][e] = (_Float16)f.x;
            pB[tb][lb + 1][e] = (_Float16)f.y;
            pB[tb][lb + 2][e] = (_Float16)f.z;
            pB[tb][lb + 3][e] = (_Float16)f.w;
            if constexpr (BI) {
                float4 q = *(const float4*)(Bi + g);
                pBi[tb][lb + 0][e] = (_Float16)q.x;
                pBi[tb][lb + 1][e] = (_Float16)q.y;
                pBi[tb][lb + 2][e] = (_Float16)q.z;
                pBi[tb][lb + 3][e] = (_Float16)q.w;
            }
        }
        if (k0 + 32 < K) {   // speculative prefetch of next K panel
            __builtin_prefetch(Ar + (size_t)(mBase + (tid & 63)) * K + k0 + 32, 0, 1);
            __builtin_prefetch(Br + (size_t)(k0 + 32 + (tid & 31)) * N + nBase, 0, 1);
        }
        __syncthreads();

        // ---- consume: vector LDS loads straight into WMMA operand regs ----
        v16h a  = *(const v16h*)&pA[wm][lane][0];
        v16h b0 = *(const v16h*)&pB[wn * 2 + 0][lane][0];
        v16h b1 = *(const v16h*)&pB[wn * 2 + 1][lane][0];

        cr0 = __builtin_amdgcn_wmma_f32_16x16x32_f16(false, a, false, b0,
                                                     (short)0, cr0, false, false);
        cr1 = __builtin_amdgcn_wmma_f32_16x16x32_f16(false, a, false, b1,
                                                     (short)0, cr1, false, false);
        if constexpr (AI && BI) {
            // pAi holds -Ai here (aSign == -1 since !CI in every such call)
            v16h an  = *(const v16h*)&pAi[wm][lane][0];
            v16h bi0 = *(const v16h*)&pBi[wn * 2 + 0][lane][0];
            v16h bi1 = *(const v16h*)&pBi[wn * 2 + 1][lane][0];
            cr0 = __builtin_amdgcn_wmma_f32_16x16x32_f16(false, an, false, bi0,
                                                         (short)0, cr0, false, false);
            cr1 = __builtin_amdgcn_wmma_f32_16x16x32_f16(false, an, false, bi1,
                                                         (short)0, cr1, false, false);
        }
        if constexpr (CI) {
            if constexpr (BI) {
                v16h bi0 = *(const v16h*)&pBi[wn * 2 + 0][lane][0];
                v16h bi1 = *(const v16h*)&pBi[wn * 2 + 1][lane][0];
                ci0 = __builtin_amdgcn_wmma_f32_16x16x32_f16(false, a, false, bi0,
                                                             (short)0, ci0, false, false);
                ci1 = __builtin_amdgcn_wmma_f32_16x16x32_f16(false, a, false, bi1,
                                                             (short)0, ci1, false, false);
            }
            if constexpr (AI) {
                v16h ap = *(const v16h*)&pAi[wm][lane][0];  // +Ai (aSign == +1 when CI)
                ci0 = __builtin_amdgcn_wmma_f32_16x16x32_f16(false, ap, false, b0,
                                                             (short)0, ci0, false, false);
                ci1 = __builtin_amdgcn_wmma_f32_16x16x32_f16(false, ap, false, b1,
                                                             (short)0, ci1, false, false);
            }
        }
        __syncthreads();
    }

    // ---- store two 16x16 f32 tiles per wave (C layout: vgpr r -> M = r + 8*half) ----
#pragma unroll
    for (int r = 0; r < 8; ++r) {
        int m = mBase + wm * 16 + r + (hh << 3);
        size_t g = (size_t)m * N + nBase + wn * 32 + l15;
        Cr[g] = cr0[r];
        Cr[g + 16] = cr1[r];
        if constexpr (CI) { Ci[g] = ci0[r]; Ci[g + 16] = ci1[r]; }
    }
}

// ---------------------------------------------------------------------------
// Fourier phase multiply:  X *= e^{-2pi i * fftfreq(kidx) * shift(cidx)}
// ---------------------------------------------------------------------------
__global__ void k_phase(const float* __restrict__ Xr, const float* __restrict__ Xi, long sIn,
                        float* __restrict__ Yr, float* __restrict__ Yi, long sOut,
                        const float* __restrict__ sc, const float* __restrict__ off,
                        int ps, int axisY) {
    int idx = blockIdx.x * blockDim.x + threadIdx.x;
    if (idx >= PIX) return;
    long b = blockIdx.z;
    const float* xr = Xr + b * sIn;
    const float* xi = Xi + b * sIn;
    float* yr = Yr + b * sOut;
    float* yi = Yi + b * sOut;

    int y = idx / NDIM, x = idx % NDIM;
    int kidx = axisY ? y : x;
    int cidx = axisY ? x : y;
    float kf = (kidx < NDIM / 2) ? (float)kidx * (1.0f / NDIM)
                                 : (float)(kidx - NDIM) * (1.0f / NDIM);
    float s   = sc[b * ps] * ((float)cidx - (NDIM - 1) * 0.5f) + off[b * ps];
    float ang = 2.0f * PI_F * kf * s;
    float c = cosf(ang), sn = sinf(ang);      // multiply by (c - i sn)
    float vr = xr[idx], vi = xi[idx];
    yr[idx] = vr * c + vi * sn;
    yi[idx] = vi * c - vr * sn;
}

// ---------------------------------------------------------------------------
// SAD over 403x403 crop for all 525 (sx,sy,theta) combos.
// Integer translation == exact circular shift of the rotated image.
// Deterministic: partials to sadPart[combo][blk], no float atomics.
// ---------------------------------------------------------------------------
__global__ __launch_bounds__(256)
void k_sad(const float* __restrict__ img1, const float* __restrict__ rot,
           float* __restrict__ sadPart, int nBlk) {
    const int CH = 403, C0 = 22;
    int combo = blockIdx.y;
    int kang = combo % 21;
    int ij   = combo / 21;
    int j    = ij % 5;       // sy index
    int i    = ij / 5;       // sx index
    int sx = i - 2, sy = j - 2;
    const float* R = rot + (size_t)kang * PIX;

    float acc = 0.0f;
    for (int t = blockIdx.x * blockDim.x + threadIdx.x; t < CH * CH;
         t += gridDim.x * blockDim.x) {
        int yy = t / CH, xx = t % CH;
        int y = C0 + yy, x = C0 + xx;
        int ys = y - sy; if (ys < 0) ys += NDIM; if (ys >= NDIM) ys -= NDIM;
        int xs = x - sx; if (xs < 0) xs += NDIM; if (xs >= NDIM) xs -= NDIM;
        acc += fabsf(img1[(size_t)y * NDIM + x] - R[(size_t)ys * NDIM + xs]);
    }
    __shared__ float red[256];
    red[threadIdx.x] = acc;
    __syncthreads();
    for (int s = 128; s > 0; s >>= 1) {
        if (threadIdx.x < s) red[threadIdx.x] += red[threadIdx.x + s];
        __syncthreads();
    }
    if (threadIdx.x == 0) sadPart[(size_t)combo * nBlk + blockIdx.x] = red[0];
}

// ---------------------------------------------------------------------------
// Reduce partials, argmin, parabolic subpixel fit; emit scalars + final-warp
// parameters (theta' = -rot_sub => a = tan(rot/2), b = -sin(rot)).
// ---------------------------------------------------------------------------
__device__ __forceinline__ float para_delta(float p0, float p1, float p2) {
    float a = 0.5f * (p0 + p2) - p1;
    float b = 0.5f * (p2 - p0);
    return -b / (2.0f * a);
}

__global__ void k_argmin_fit(const float* __restrict__ sadPart, int nBlk,
                             float* __restrict__ sad,      // [525] scratch
                             float* __restrict__ out,      // d_out[0..2]
                             float* __restrict__ fpar) {   // [4] final warp params
    int t = threadIdx.x;
    if (t < 525) {
        float s = 0.0f;
        for (int q = 0; q < nBlk; ++q) s += sadPart[(size_t)t * nBlk + q];
        sad[t] = s;
    }
    __syncthreads();
    if (t != 0) return;

    int best = 0; float bv = sad[0];
    for (int u = 1; u < 525; ++u) if (sad[u] < bv) { bv = sad[u]; best = u; }
    int k = best % 21, ij = best / 21, j = ij % 5, i = ij / 5;

    #define S3(a_, b_, c_) sad[(((a_) * 5) + (b_)) * 21 + (c_)]
    float dx = para_delta(S3((i + 4) % 5, j, k), S3(i, j, k), S3((i + 1) % 5, j, k));
    float dy = para_delta(S3(i, (j + 4) % 5, k), S3(i, j, k), S3(i, (j + 1) % 5, k));
    float dr = para_delta(S3(i, j, (k + 20) % 21), S3(i, j, k), S3(i, j, (k + 1) % 21));
    #undef S3

    float sx_sub  = (float)(i - 2) + dx;
    float sy_sub  = (float)(j - 2) + dy;
    float rot_sub = ((float)(k - 10) + dr) * (PI_F / 180.0f);

    out[0] = sx_sub; out[1] = sy_sub; out[2] = rot_sub;

    fpar[0] = tanf(0.5f * rot_sub);   // a = -tan(-rot/2)
    fpar[1] = -sinf(rot_sub);         // b = sin(-rot)
    fpar[2] = -sx_sub;                // x translation
    fpar[3] = -sy_sub;                // y translation
}

// ---------------------------------------------------------------------------
// Host-side orchestration (all launches on `stream`, graph-capture safe).
// ---------------------------------------------------------------------------
extern "C" void kernel_launch(void* const* d_in, const int* in_sizes, int n_in,
                              void* d_out, int out_size, void* d_ws, size_t ws_size,
                              hipStream_t stream) {
    const float* img1 = (const float*)d_in[0];
    const float* img2 = (const float*)d_in[1];
    float* out = (float*)d_out;
    float* W   = (float*)d_ws;

    const long P = PIX;
    float *Fr = W,           *Fi = W + P,      *Gr = W + 2 * P, *Gi = W + 3 * P;
    float *S0r = W + 4 * P,  *S0i = W + 5 * P;
    float *Xr = W + 6 * P;                 // [21][P]
    float *Xi = W + 27 * P;                // [21][P]
    float *T1 = W + 48 * P;                // [21][P]
    float *T2 = W + 69 * P;                // [21][P]
    float *par = W + 90 * P;
    float *prot_a = par, *prot_b = par + 32, *zer = par + 64, *fpar = par + 96;
    float *sad = par + 128;                // [544]
    float *sadPart = par + 704;            // [525 * 64]

    const int SADBLK = 64;
    dim3 blk(256);
    dim3 gg1(NDIM / 64, NDIM / 64, 1);
    dim3 gg21(NDIM / 64, NDIM / 64, 21);
    dim3 pe1((PIX + 255) / 256, 1, 1);
    dim3 pe21((PIX + 255) / 256, 1, 21);

    // GEMM flavors:
    //   fwdR : real x complex -> complex   (FFT of a real image, right-multiply)
    //   invC : complex x complex -> real   (inverse FFT keeping .real)
    //   fwdL : complex x real -> complex   (left-multiplied FFT: F * img)
    auto fwdR = k_gemm_cplx<false, true, true>;
    auto invC = k_gemm_cplx<true,  true, false>;
    auto fwdL = k_gemm_cplx<true,  false, true>;

    k_build_dft<<<(PIX + 255) / 256, blk, 0, stream>>>(Fr, Fi, Gr, Gi);
    k_init<<<1, 64, 0, stream>>>(prot_a, prot_b, zer);

    // ---- 21 rotations: shear_x(a*yy) o shear_y(b*xx) o shear_x(a*yy) ----
    fwdR<<<gg1, blk, 0, stream>>>(img2, nullptr, 0, Fr, Fi, 0, S0r, S0i, 0);
    k_phase<<<pe21, blk, 0, stream>>>(S0r, S0i, 0, Xr, Xi, P, prot_a, zer, 1, 0);
    invC<<<gg21, blk, 0, stream>>>(Xr, Xi, P, Gr, Gi, 0, T1, nullptr, P);

    fwdL<<<gg21, blk, 0, stream>>>(Fr, Fi, 0, T1, nullptr, P, Xr, Xi, P);
    k_phase<<<pe21, blk, 0, stream>>>(Xr, Xi, P, Xr, Xi, P, prot_b, zer, 1, 1);
    invC<<<gg21, blk, 0, stream>>>(Gr, Gi, 0, Xr, Xi, P, T2, nullptr, P);

    fwdR<<<gg21, blk, 0, stream>>>(T2, nullptr, P, Fr, Fi, 0, Xr, Xi, P);
    k_phase<<<pe21, blk, 0, stream>>>(Xr, Xi, P, Xr, Xi, P, prot_a, zer, 1, 0);
    invC<<<gg21, blk, 0, stream>>>(Xr, Xi, P, Gr, Gi, 0, T1, nullptr, P);
    // T1 = rot[21]  (16.9 MB, L2-resident)

    // ---- SAD over 525 combos (integer shifts == circular shifts) ----
    dim3 gsad(SADBLK, 525);
    k_sad<<<gsad, blk, 0, stream>>>(img1, T1, sadPart, SADBLK);
    k_argmin_fit<<<1, 544, 0, stream>>>(sadPart, SADBLK, sad, out, fpar);

    // ---- Final subpixel warp of img2: 3 rotation shears + 2 translation shears ----
    k_phase<<<pe1, blk, 0, stream>>>(S0r, S0i, 0, Xr, Xi, 0, &fpar[0], zer, 0, 0);
    invC<<<gg1, blk, 0, stream>>>(Xr, Xi, 0, Gr, Gi, 0, T2, nullptr, 0);

    fwdL<<<gg1, blk, 0, stream>>>(Fr, Fi, 0, T2, nullptr, 0, Xr, Xi, 0);
    k_phase<<<pe1, blk, 0, stream>>>(Xr, Xi, 0, Xr, Xi, 0, &fpar[1], zer, 0, 1);
    invC<<<gg1, blk, 0, stream>>>(Gr, Gi, 0, Xr, Xi, 0, T1, nullptr, 0);

    fwdR<<<gg1, blk, 0, stream>>>(T1, nullptr, 0, Fr, Fi, 0, Xr, Xi, 0);
    k_phase<<<pe1, blk, 0, stream>>>(Xr, Xi, 0, Xr, Xi, 0, &fpar[0], zer, 0, 0);
    invC<<<gg1, blk, 0, stream>>>(Xr, Xi, 0, Gr, Gi, 0, T2, nullptr, 0);

    fwdR<<<gg1, blk, 0, stream>>>(T2, nullptr, 0, Fr, Fi, 0, Xr, Xi, 0);
    k_phase<<<pe1, blk, 0, stream>>>(Xr, Xi, 0, Xr, Xi, 0, zer, &fpar[2], 0, 0);
    invC<<<gg1, blk, 0, stream>>>(Xr, Xi, 0, Gr, Gi, 0, T1, nullptr, 0);

    fwdL<<<gg1, blk, 0, stream>>>(Fr, Fi, 0, T1, nullptr, 0, Xr, Xi, 0);
    k_phase<<<pe1, blk, 0, stream>>>(Xr, Xi, 0, Xr, Xi, 0, zer, &fpar[3], 0, 1);
    invC<<<gg1, blk, 0, stream>>>(Gr, Gi, 0, Xr, Xi, 0, out + 3, nullptr, 0);
}
